// MultiHeadAttention_37288906064252
// MI455X (gfx1250) — matrix operand order repaired and checked by reading, not compiled, not run on hardware
//
#include <hip/hip_runtime.h>
#include <hip/hip_bf16.h>
#include <math.h>

// Problem constants (from reference)
#define Bq   4
#define Sq   2048
#define Fq   64
#define HIDq 256
#define Dq   128
#define Hq   8
#define DKq  16

// softmax uses exp2; Q is pre-scaled by (1/sqrt(dk)) * log2(e)
#define QSCALE 0.3606737602222409f

// WMMA fragment types (wave32, V_WMMA_F32_16X16X32_F16 layouts, ISA 7.12.2)
typedef __attribute__((ext_vector_type(16))) _Float16 v16h;
typedef __attribute__((ext_vector_type(8)))  _Float16 v8h;
typedef __attribute__((ext_vector_type(8)))  float    v8f;
union AF { v16h v; v8h h8[2]; _Float16 h[16]; };  // A(16x32)/B(32x16) f16 frag
union CF { v8f  v; float f[8]; };                 // C/D (16x16) f32 frag

// workspace layout (in _Float16 units)
// Q,K: [B,H,S,DK] row-major.  V: stored TRANSPOSED [B,H,DK,S].
#define WS_Q   0
#define WS_K   (Bq*Hq*Sq*DKq)
#define WS_VT  (2*Bq*Hq*Sq*DKq)
#define WS_W1T (3*Bq*Hq*Sq*DKq)             // 3 x W1^T [HID][F]
#define WS_W2T (WS_W1T + 3*Fq*HIDq)         // 3 x W2^T [D][HID]

__device__ __forceinline__ void atomicMaxF(float* addr, float v) {
    if (v >= 0.0f) atomicMax((int*)addr, __float_as_int(v));
    else           atomicMin((unsigned int*)addr, __float_as_uint(v));
}

__global__ void init_out(float* out, int n) {
    int i = blockIdx.x * blockDim.x + threadIdx.x;
    if (i < n) out[i] = -INFINITY;
}

// f32 -> f16 with transpose so GEMM B-fragments are lane-contiguous.
__global__ void convert_weights(const float* qW1, const float* qW2,
                                const float* kW1, const float* kW2,
                                const float* vW1, const float* vW2,
                                _Float16* ws) {
    int i = blockIdx.x * blockDim.x + threadIdx.x;
    const int n1 = Fq * HIDq;   // 16384
    const int n2 = HIDq * Dq;   // 32768
    if (i < n1) {               // W1[k][n] -> W1t[n][k]
        int k = i / HIDq, n = i % HIDq;
        int t = n * Fq + k;
        ws[WS_W1T + 0 * n1 + t] = (_Float16)qW1[i];
        ws[WS_W1T + 1 * n1 + t] = (_Float16)kW1[i];
        ws[WS_W1T + 2 * n1 + t] = (_Float16)vW1[i];
    }
    if (i < n2) {               // W2[k][n] -> W2t[n][k]
        int k = i / Dq, n = i % Dq;
        int t = n * HIDq + k;
        ws[WS_W2T + 0 * n2 + t] = (_Float16)qW2[i];
        ws[WS_W2T + 1 * n2 + t] = (_Float16)kW2[i];
        ws[WS_W2T + 2 * n2 + t] = (_Float16)vW2[i];
    }
}

// One wave per 16-row tile per projection:
//   h = elu(x @ W1 + b1) [16x256] -> LDS,  out = h @ W2 + b2 [16x128]
__global__ __launch_bounds__(32)
void qkv_proj(const float* __restrict__ x,
              const float* __restrict__ b1q, const float* __restrict__ b2q,
              const float* __restrict__ b1k, const float* __restrict__ b2k,
              const float* __restrict__ b1v, const float* __restrict__ b2v,
              _Float16* __restrict__ ws) {
    __shared__ _Float16 hlds[16][HIDq];           // 8 KB ELU(hidden) tile
    const int lane = threadIdx.x & 31;
    const int half = lane >> 4;
    const int lm   = lane & 15;
    const int tile = blockIdx.x;                  // 0..511
    const int proj = blockIdx.y;                  // 0=q 1=k 2=v
    const int row0 = tile * 16;

    const _Float16* W1t = ws + WS_W1T + proj * (Fq * HIDq);   // [256][64]
    const _Float16* W2t = ws + WS_W2T + proj * (HIDq * Dq);   // [128][256]
    const float* b1 = (proj == 0) ? b1q : (proj == 1) ? b1k : b1v;
    const float* b2 = (proj == 0) ? b2q : (proj == 1) ? b2k : b2v;
    const float oscale = (proj == 0) ? QSCALE : 1.0f;   // fold softmax scale

    // x tile as two A fragments; contiguous 8-float chunks -> float4 loads
    const float* xrow = x + (size_t)(row0 + lm) * Fq;
    AF xa[2];
    for (int ks = 0; ks < 2; ++ks)
        for (int c = 0; c < 2; ++c) {
            int k0 = ks * 32 + 16 * half + 8 * c;
            float4 f0 = *(const float4*)(xrow + k0);
            float4 f1 = *(const float4*)(xrow + k0 + 4);
            xa[ks].h[8 * c + 0] = (_Float16)f0.x;
            xa[ks].h[8 * c + 1] = (_Float16)f0.y;
            xa[ks].h[8 * c + 2] = (_Float16)f0.z;
            xa[ks].h[8 * c + 3] = (_Float16)f0.w;
            xa[ks].h[8 * c + 4] = (_Float16)f1.x;
            xa[ks].h[8 * c + 5] = (_Float16)f1.y;
            xa[ks].h[8 * c + 6] = (_Float16)f1.z;
            xa[ks].h[8 * c + 7] = (_Float16)f1.w;
        }

    // GEMM1: [16x64] @ [64x256] + b1, ELU -> LDS
    for (int nt = 0; nt < HIDq / 16; ++nt) {
        CF c;
        float bias = b1[nt * 16 + lm];
        for (int r = 0; r < 8; ++r) c.f[r] = bias;
        const _Float16* wrow = W1t + (size_t)(nt * 16 + lm) * Fq;
        for (int ks = 0; ks < 2; ++ks) {
            AF bf;                                    // k = ks*32+16*half..+15
            bf.h8[0] = *(const v8h*)(wrow + ks * 32 + 16 * half);
            bf.h8[1] = *(const v8h*)(wrow + ks * 32 + 16 * half + 8);
            c.v = __builtin_amdgcn_wmma_f32_16x16x32_f16(
                false, xa[ks].v, false, bf.v, (short)0, c.v, false, false);
        }
        for (int r = 0; r < 8; ++r) {
            float hv = c.f[r];
            hv = hv > 0.0f ? hv : (__expf(hv) - 1.0f);   // ELU
            hlds[r + 8 * half][nt * 16 + lm] = (_Float16)hv;
        }
    }

    // GEMM2: [16x256] @ [256x128] + b2 -> Q (prescaled) / K / V^T
    for (int nt = 0; nt < Dq / 16; ++nt) {            // nt == head (DK==16)
        CF c;
        float bias = b2[nt * 16 + lm];
        for (int r = 0; r < 8; ++r) c.f[r] = bias;
        const _Float16* wrow = W2t + (size_t)(nt * 16 + lm) * HIDq;
        for (int ks = 0; ks < HIDq / 32; ++ks) {
            AF af, bf;
            int kb = ks * 32 + 8 * half;
            af.h8[0] = *(const v8h*)&hlds[lm][kb];
            af.h8[1] = *(const v8h*)&hlds[lm][kb + 16];
            bf.h8[0] = *(const v8h*)(wrow + ks * 32 + 8 * half);
            bf.h8[1] = *(const v8h*)(wrow + ks * 32 + 16 + 8 * half);
            c.v = __builtin_amdgcn_wmma_f32_16x16x32_f16(
                false, af.v, false, bf.v, (short)0, c.v, false, false);
        }
        for (int r = 0; r < 8; ++r) {
            int row = row0 + r + 8 * half;
            int b = row / Sq, s = row % Sq;
            size_t bh = (size_t)b * Hq + nt;
            float val = c.f[r] * oscale;
            if (proj == 2)       // V^T: [B,H,DK,S]
                ws[WS_VT + (bh * DKq + lm) * Sq + s] = (_Float16)val;
            else if (proj == 1)  // K: [B,H,S,DK]
                ws[WS_K + (bh * Sq + s) * DKq + lm] = (_Float16)val;
            else                 // Q (pre-scaled by QSCALE)
                ws[WS_Q + (bh * Sq + s) * DKq + lm] = (_Float16)val;
        }
    }
}

// Flash attention, 32 keys x 32 queries per wave per step:
//   4 padded QK^T WMMAs + 2 rowsum WMMAs (P @ ones) + 2 dense PV WMMAs.
// K/V^T fragments are shared across the two query tiles.
__global__ __launch_bounds__(128)
void attention_max(const _Float16* __restrict__ ws, float* __restrict__ outMax) {
    __shared__ _Float16 plds[4][2][16][32];       // per-wave, per-qtile P
    const int lane = threadIdx.x & 31;
    const int wid  = threadIdx.x >> 5;
    const int half = lane >> 4;
    const int lm   = lane & 15;
    const int w  = blockIdx.x * 4 + wid;          // 0..2047
    const int qp = w & 63;                        // 32-query block
    const int h  = (w >> 6) & 7;
    const int b  = w >> 9;

    const _Float16* Q  = ws + WS_Q;
    const _Float16* K  = ws + WS_K;
    const _Float16* Vt = ws + WS_VT;
    const size_t base  = ((size_t)b * Hq + h) * Sq;
    const _Float16* vrow = Vt + (base * DKq) + (size_t)lm * Sq; // V^T row dk=lm

    // ones B-fragment for rowsum-by-WMMA
    AF ones;
    for (int i = 0; i < 16; ++i) ones.h[i] = (_Float16)1.0f;

    // Two Q A-fragments (upper-half K padding = 0)
    AF qa[2];
    for (int t2 = 0; t2 < 2; ++t2) {
        qa[t2].h8[0] = *(const v8h*)(Q + (base + qp * 32 + t2 * 16 + lm) * DKq
                                       + 8 * half);
        for (int i = 8; i < 16; ++i) qa[t2].h[i] = (_Float16)0.0f;
    }

    CF acc[2];
    float mrow[2][8], lrow[2][8];
    for (int t2 = 0; t2 < 2; ++t2)
        for (int r = 0; r < 8; ++r) {
            acc[t2].f[r] = 0.0f; mrow[t2][r] = -INFINITY; lrow[t2][r] = 0.0f;
        }

    for (int kt = 0; kt < Sq / 32; ++kt) {
        const int key0 = kt * 32;
        if (kt + 1 < Sq / 32) {
            __builtin_prefetch(K + (base + key0 + 32 + lm) * DKq, 0, 1);
            __builtin_prefetch(vrow + key0 + 32, 0, 1);
        }

        // shared K^T B-fragments (two 16-key tiles, dk padded 16->32)
        AF kb[2];
        for (int t = 0; t < 2; ++t) {
            const _Float16* krow = K + (base + key0 + t * 16 + lm) * DKq;
            kb[t].h8[0] = *(const v8h*)(krow + 8 * half);
            for (int i = 8; i < 16; ++i) kb[t].h[i] = (_Float16)0.0f;
        }
        // shared V^T B-fragment (dense 32x16)
        AF vb;
    vb.h8[0] = *(const v8h*)(vrow + key0 + 8 * half);
    vb.h8[1] = *(const v8h*)(vrow + key0 + 16 + 8 * half);

        for (int t2 = 0; t2 < 2; ++t2) {
            v8f z = {};
            CF sc0, sc1;
            sc0.v = __builtin_amdgcn_wmma_f32_16x16x32_f16(
                false, qa[t2].v, false, kb[0].v, (short)0, z, false, false);
            sc1.v = __builtin_amdgcn_wmma_f32_16x16x32_f16(
                false, qa[t2].v, false, kb[1].v, (short)0, z, false, false);

            // online softmax (base-2; scores already scaled by QSCALE)
            for (int r = 0; r < 8; ++r) {
                float s0 = sc0.f[r], s1 = sc1.f[r];
                float rm = fmaxf(s0, s1);
                rm = fmaxf(rm, __shfl_xor(rm, 1, 32));
                rm = fmaxf(rm, __shfl_xor(rm, 2, 32));
                rm = fmaxf(rm, __shfl_xor(rm, 4, 32));
                rm = fmaxf(rm, __shfl_xor(rm, 8, 32));
                float mnew  = fmaxf(mrow[t2][r], rm);
                float alpha = exp2f(mrow[t2][r] - mnew);
                float p0    = exp2f(s0 - mnew);
                float p1    = exp2f(s1 - mnew);
                mrow[t2][r]  = mnew;
                lrow[t2][r] *= alpha;
                acc[t2].f[r] *= alpha;
                plds[wid][t2][r + 8 * half][lm]      = (_Float16)p0;
                plds[wid][t2][r + 8 * half][lm + 16] = (_Float16)p1;
            }

            // P as dense 16x32 A fragment (2x ds b128; same-wave DS in-order)
            AF pa;
            pa.h8[0] = *(const v8h*)&plds[wid][t2][lm][8 * half];
            pa.h8[1] = *(const v8h*)&plds[wid][t2][lm][16 + 8 * half];

            // rowsum(P) via WMMA with ones; C-layout matches lrow indexing
            CF sums;
            sums.v = __builtin_amdgcn_wmma_f32_16x16x32_f16(
                false, pa.v, false, ones.v, (short)0, z, false, false);
            for (int r = 0; r < 8; ++r) lrow[t2][r] += sums.f[r];

            acc[t2].v = __builtin_amdgcn_wmma_f32_16x16x32_f16(
                false, pa.v, false, vb.v, (short)0, acc[t2].v, false, false);
        }
    }

    // finalize (/l), column max over 32 rows, merge halves, atomic into out
    float pm = -INFINITY;
    for (int t2 = 0; t2 < 2; ++t2)
        for (int r = 0; r < 8; ++r)
            pm = fmaxf(pm, acc[t2].f[r] / lrow[t2][r]);
    pm = fmaxf(pm, __shfl_xor(pm, 16, 32));
    if (half == 0)
        atomicMaxF(outMax + b * Dq + h * DKq + lm, pm);
}

extern "C" void kernel_launch(void* const* d_in, const int* in_sizes, int n_in,
                              void* d_out, int out_size, void* d_ws, size_t ws_size,
                              hipStream_t stream) {
    (void)in_sizes; (void)n_in; (void)out_size; (void)ws_size;
    const float* x   = (const float*)d_in[0];
    const float* qW1 = (const float*)d_in[1];
    const float* qb1 = (const float*)d_in[2];
    const float* qW2 = (const float*)d_in[3];
    const float* qb2 = (const float*)d_in[4];
    const float* kW1 = (const float*)d_in[5];
    const float* kb1 = (const float*)d_in[6];
    const float* kW2 = (const float*)d_in[7];
    const float* kb2 = (const float*)d_in[8];
    const float* vW1 = (const float*)d_in[9];
    const float* vb1 = (const float*)d_in[10];
    const float* vW2 = (const float*)d_in[11];
    const float* vb2 = (const float*)d_in[12];
    _Float16* ws = (_Float16*)d_ws;     // ~6.6 MB used
    float* out = (float*)d_out;         // [4,128]

    init_out<<<1, Bq * Dq, 0, stream>>>(out, Bq * Dq);
    convert_weights<<<(HIDq * Dq + 255) / 256, 256, 0, stream>>>(
        qW1, qW2, kW1, kW2, vW1, vW2, ws);
    qkv_proj<<<dim3(Bq * Sq / 16, 3), 32, 0, stream>>>(
        x, qb1, qb2, kb1, kb2, vb1, vb2, ws);
    attention_max<<<Bq * Hq * (Sq / 32) / 4, 128, 0, stream>>>(ws, out);
}
